// MultiHeadAttention_39865886441911
// MI455X (gfx1250) — matrix operand order repaired and checked by reading, not compiled, not run on hardware
//
#include <hip/hip_runtime.h>
#include <hip/hip_bf16.h>

typedef __attribute__((ext_vector_type(16))) _Float16 v16h;
typedef __attribute__((ext_vector_type(8)))  float    v8f;

#define SEQ    2048
#define DMODEL 2048
#define NH     32
#define NKV    8
#define HD     64

union Frag16 { v16h v; unsigned u[8]; };

__device__ __forceinline__ v8f wmma_f16(const Frag16& a, const Frag16& b, v8f c) {
    // D = A(16x32 f16) * B(32x16 f16) + C(16x16 f32)
    return __builtin_amdgcn_wmma_f32_16x16x32_f16(
        false, a.v, false, b.v, (short)0, c, false, false);
}

// ---- CDNA5 async DMA: global -> LDS, tracked by ASYNCcnt (GVS addressing) ----
__device__ __forceinline__ void async_ld_b128(unsigned lds_addr, const void* base,
                                              unsigned byte_off) {
    asm volatile("global_load_async_to_lds_b128 %0, %1, %2"
                 :: "v"(lds_addr), "v"(byte_off), "s"(base) : "memory");
}
__device__ __forceinline__ void wait_async0() {
    asm volatile("s_wait_asynccnt 0" ::: "memory");
}
// flat shared-aperture address keeps the LDS byte offset in its low 32 bits
__device__ __forceinline__ unsigned lds_off(const void* p) {
    return (unsigned)(unsigned long long)p;
}

// ---- DPP16 butterfly reductions within each 16-lane row (VALU speed) ----
template <int CTRL>
__device__ __forceinline__ float dpp_mov(float x) {
    union { float f; int i; } s, r; s.f = x;
    r.i = __builtin_amdgcn_update_dpp(s.i, s.i, CTRL, 0xF, 0xF, true);
    return r.f;
}
__device__ __forceinline__ float row_max16(float x) {
    x = fmaxf(x, dpp_mov<0xB1>(x));    // quad_perm [1,0,3,2]  (xor 1)
    x = fmaxf(x, dpp_mov<0x4E>(x));    // quad_perm [2,3,0,1]  (xor 2)
    x = fmaxf(x, dpp_mov<0x141>(x));   // row_half_mirror
    x = fmaxf(x, dpp_mov<0x140>(x));   // row_mirror
    return x;
}
__device__ __forceinline__ float row_sum16(float x) {
    x += dpp_mov<0xB1>(x);
    x += dpp_mov<0x4E>(x);
    x += dpp_mov<0x141>(x);
    x += dpp_mov<0x140>(x);
    return x;
}

// ---------------------------------------------------------------------------
// Precision conversion passes (bandwidth-trivial at 23.3 TB/s)
// ---------------------------------------------------------------------------
__global__ void f32_to_f16(const float* __restrict__ in, _Float16* __restrict__ out) {
    int i = blockIdx.x * blockDim.x + threadIdx.x;
    out[i] = (_Float16)in[i];
}

// W[K][N] f32 -> Wt[N][K] f16 (LDS-tiled transpose)
__global__ __launch_bounds__(256)
void transpose_to_f16(const float* __restrict__ in, _Float16* __restrict__ out,
                      int K, int N) {
    __shared__ float tile[32][33];
    int k0 = blockIdx.x * 32, n0 = blockIdx.y * 32;
    for (int i = threadIdx.y; i < 32; i += 8)
        tile[i][threadIdx.x] = in[(size_t)(k0 + i) * N + n0 + threadIdx.x];
    __syncthreads();
    for (int i = threadIdx.y; i < 32; i += 8)
        out[(size_t)(n0 + i) * K + k0 + threadIdx.x] = (_Float16)tile[threadIdx.x][i];
}

// ---------------------------------------------------------------------------
// f16 GEMM, 128x128 block tile, 256 thr = 8 waves, each wave 32x64 (8 C tiles).
// K-step 64: 16 WMMAs per wave between barriers. Double-buffered async-DMA
// staging overlaps tile t+1's DMA with tile t's WMMAs. Epilogue fuses RoPE +
// head relayout (rotate-half partner t^32 lives in col-frag nt^2, same lane).
// ---------------------------------------------------------------------------
enum EpiMode { EPI_F32, EPI_ROPE, EPI_COPY_T };

template <EpiMode MODE>
__global__ __launch_bounds__(256)
void gemm_f16_async(const _Float16* __restrict__ A, const _Float16* __restrict__ Bt,
                    void* __restrict__ Out, int M, int N, int K, int nheads) {
    __shared__ _Float16 As[2][128 * 64];
    __shared__ _Float16 Bs[2][128 * 64];

    const int tid   = threadIdx.x;
    const int lane  = tid & 31;
    const int w     = tid >> 5;       // 0..7
    const int wr    = w >> 1;         // 0..3 : 32-row slice
    const int wc    = w & 1;          // 0..1 : 64-col slice
    const int mlane = lane & 15;
    const int half  = lane >> 4;
    const int m0    = blockIdx.y * 128;
    const int n0    = blockIdx.x * 128;

    v8f acc[2][4] = {};

    auto stage = [&](int buf, int k0) {
        // 1024 x 16B chunks per matrix per tile; each thread issues 4 A + 4 B
#pragma unroll
        for (int j = 0; j < 4; ++j) {
            int c = tid + 256 * j;
            int row = c >> 3, part = c & 7;
            async_ld_b128(lds_off(&As[buf][row * 64 + part * 8]), A,
                          (unsigned)(((m0 + row) * K + k0 + part * 8) * 2));
            async_ld_b128(lds_off(&Bs[buf][row * 64 + part * 8]), Bt,
                          (unsigned)(((n0 + row) * K + k0 + part * 8) * 2));
        }
    };

    const int nT = K / 64;
    stage(0, 0);
    wait_async0();
    __syncthreads();

    int buf = 0;
    for (int t = 0; t < nT; ++t) {
        if (t + 1 < nT) stage(buf ^ 1, (t + 1) * 64);  // overlap with WMMAs below

#pragma unroll
        for (int kc = 0; kc < 2; ++kc) {               // two 32-wide K chunks
            Frag16 af[2];
#pragma unroll
            for (int rf = 0; rf < 2; ++rf)
#pragma unroll
                for (int v = 0; v < 8; ++v) {
                    int ka = 2 * v + 16 * half;
                    af[rf].u[v] = *(const unsigned*)
                        &As[buf][(wr * 32 + rf * 16 + mlane) * 64 + kc * 32 + ka];
                }
#pragma unroll
            for (int nt = 0; nt < 4; ++nt) {
                Frag16 bf;
#pragma unroll
                for (int v = 0; v < 8; ++v)
                    bf.u[v] = *(const unsigned*)
                        &Bs[buf][(wc * 64 + nt * 16 + mlane) * 64 + kc * 32 + half * 16 + 2 * v];
                acc[0][nt] = wmma_f16(af[0], bf, acc[0][nt]);
                acc[1][nt] = wmma_f16(af[1], bf, acc[1][nt]);
            }
        }
        wait_async0();      // tile t+1 landed (issued before compute -> overlapped)
        __syncthreads();
        buf ^= 1;
    }

    // ---- epilogue ----
    const int h = (n0 + wc * 64) >> 6;   // this wave's head (HD == 64)
#pragma unroll
    for (int rf = 0; rf < 2; ++rf)
#pragma unroll
        for (int v = 0; v < 8; ++v) {
            int m = m0 + wr * 32 + rf * 16 + v + 8 * half;
            if (MODE == EPI_F32) {
                float* O = (float*)Out;
#pragma unroll
                for (int nt = 0; nt < 4; ++nt)
                    O[(size_t)m * N + n0 + wc * 64 + nt * 16 + mlane] = acc[rf][nt][v];
            } else {
                _Float16* O = (_Float16*)Out;
                int b = m >> 11, s = m & (SEQ - 1);
                if (MODE == EPI_ROPE) {
#pragma unroll
                    for (int nt = 0; nt < 4; ++nt) {
                        int tcol = nt * 16 + mlane;
                        int j    = tcol & 31;
                        float inv = __expf(-0.2878231366242557f * (float)j); // ln(1e4)/32
                        float ang = (float)s * inv;
                        float cs = __cosf(ang), sn = __sinf(ang);
                        float xv = acc[rf][nt][v];
                        float xp = acc[rf][nt ^ 2][v];          // rotate-half partner
                        float y  = xv * cs + ((tcol < 32) ? -xp : xp) * sn;
                        O[(((size_t)b * nheads + h) * SEQ + s) * HD + tcol] = (_Float16)y;
                    }
                } else {  // EPI_COPY_T : V relayout to [B][Hkv][HD][SEQ]
#pragma unroll
                    for (int nt = 0; nt < 4; ++nt) {
                        int tcol = nt * 16 + mlane;
                        O[(((size_t)b * nheads + h) * HD + tcol) * SEQ + s] =
                            (_Float16)acc[rf][nt][v];
                    }
                }
            }
        }
}

// ---------------------------------------------------------------------------
// Flash attention (causal, GQA 4:1): block = 128 thr = 4 waves = 64 queries.
// Double-buffered async-DMA staged K [32key][64d] / V [64d][32key] tiles,
// 4 WMMAs QK^T, DPP online softmax (mask block skipped on interior tiles),
// P->A-frag via per-wave LDS, 4 WMMAs PV.
// ---------------------------------------------------------------------------
__global__ __launch_bounds__(128)
void flash_attn(const _Float16* __restrict__ Q, const _Float16* __restrict__ Kb,
                const _Float16* __restrict__ VbT, _Float16* __restrict__ Out) {
    __shared__ _Float16 Ks[2][32 * 64];
    __shared__ _Float16 Vs[2][64 * 32];
    __shared__ _Float16 Ps[4][16 * 32];

    const int tid   = threadIdx.x;
    const int lane  = tid & 31;
    const int w     = tid >> 5;
    const int mlane = lane & 15;
    const int half  = lane >> 4;

    const int blk = blockIdx.x;
    const int qt  = blk & 31;
    const int h   = (blk >> 5) & (NH - 1);
    const int b   = blk >> 10;
    const int hkv = h >> 2;

    const int q0 = qt * 64;
    const int qw = q0 + w * 16;

    const _Float16* Qp = Q   + (((size_t)b * NH  + h)   * SEQ + qw) * HD;
    const _Float16* Kp = Kb  + (((size_t)b * NKV + hkv) * SEQ) * HD;
    const _Float16* Vp = VbT + (((size_t)b * NKV + hkv) * HD) * SEQ;

    Frag16 qf0, qf1;
#pragma unroll
    for (int v = 0; v < 8; ++v) {
        int ka = 2 * v + 16 * half;
        qf0.u[v] = *(const unsigned*)(Qp + mlane * HD + ka);
        qf1.u[v] = *(const unsigned*)(Qp + mlane * HD + 32 + ka);
    }

    float m_i[8], l_i[8];
    v8f oacc[4] = {};
#pragma unroll
    for (int v = 0; v < 8; ++v) { m_i[v] = -1e30f; l_i[v] = 0.0f; }

    auto stage = [&](int buf, int kbase) {
#pragma unroll
        for (int j = 0; j < 2; ++j) {               // K tile: 256 x 16B chunks
            int c = tid + 128 * j;
            int key = c >> 3, part = c & 7;
            async_ld_b128(lds_off(&Ks[buf][key * 64 + part * 8]), Kp,
                          (unsigned)((((kbase + key) * HD) + part * 8) * 2));
        }
#pragma unroll
        for (int j = 0; j < 2; ++j) {               // V tile: 256 x 16B chunks
            int c = tid + 128 * j;
            int d = c >> 2, part = c & 3;
            async_ld_b128(lds_off(&Vs[buf][d * 32 + part * 8]), Vp,
                          (unsigned)(((d * SEQ) + kbase + part * 8) * 2));
        }
    };

    const int nkt = q0 / 32 + 2;
    stage(0, 0);
    wait_async0();
    __syncthreads();

    int buf = 0;
    for (int t = 0; t < nkt; ++t) {
        const int kbase = t * 32;
        if (t + 1 < nkt) stage(buf ^ 1, kbase + 32);   // overlap with compute

        // ---- S = Q K^T ----
        v8f s0 = {}, s1 = {};
        Frag16 bk;
#pragma unroll
        for (int dc = 0; dc < 2; ++dc) {
#pragma unroll
            for (int v = 0; v < 8; ++v)
                bk.u[v] = *(const unsigned*)&Ks[buf][mlane * 64 + dc * 32 + half * 16 + 2 * v];
            s0 = wmma_f16(dc ? qf1 : qf0, bk, s0);
#pragma unroll
            for (int v = 0; v < 8; ++v)
                bk.u[v] = *(const unsigned*)&Ks[buf][(16 + mlane) * 64 + dc * 32 + half * 16 + 2 * v];
            s1 = wmma_f16(dc ? qf1 : qf0, bk, s1);
        }

        // ---- scale + (diagonal-only) causal mask + online softmax ----
        const bool need_mask = (kbase + 31 > qw);   // wave-uniform -> s_cbranch
        float p0[8], p1[8];
#pragma unroll
        for (int v = 0; v < 8; ++v) {
            float sv0 = s0[v] * 0.125f;
            float sv1 = s1[v] * 0.125f;
            if (need_mask) {
                int q = qw + v + 8 * half;
                if (kbase + mlane > q)      sv0 = -1e30f;
                if (kbase + 16 + mlane > q) sv1 = -1e30f;
            }
            float mx   = row_max16(fmaxf(sv0, sv1));
            float mnew = fmaxf(m_i[v], mx);
            float corr = __expf(m_i[v] - mnew);
            m_i[v] = mnew;
            float e0 = __expf(sv0 - mnew);
            float e1 = __expf(sv1 - mnew);
            float rs = row_sum16(e0 + e1);
            l_i[v] = l_i[v] * corr + rs;
#pragma unroll
            for (int nt = 0; nt < 4; ++nt) oacc[nt][v] *= corr;
            p0[v] = e0; p1[v] = e1;
        }

        // ---- C-layout P -> A-fragment via per-wave LDS round-trip ----
        _Float16* Pw = &Ps[w][0];
#pragma unroll
        for (int v = 0; v < 8; ++v) {
            int r = v + 8 * half;
            Pw[r * 32 + mlane]      = (_Float16)p0[v];
            Pw[r * 32 + 16 + mlane] = (_Float16)p1[v];
        }
        asm volatile("s_wait_dscnt 0" ::: "memory");
        Frag16 pf;
#pragma unroll
        for (int v = 0; v < 8; ++v) {
            int ka = 2 * v + 16 * half;
            pf.u[v] = *(const unsigned*)&Pw[mlane * 32 + ka];
        }

        // ---- O += P V ----
        Frag16 bv;
#pragma unroll
        for (int nt = 0; nt < 4; ++nt) {
#pragma unroll
            for (int v = 0; v < 8; ++v)
                bv.u[v] = *(const unsigned*)&Vs[buf][(nt * 16 + mlane) * 32 + half * 16 + 2 * v];
            oacc[nt] = wmma_f16(pf, bv, oacc[nt]);
        }
        wait_async0();
        __syncthreads();
        buf ^= 1;
    }

    // ---- finalize ----
#pragma unroll
    for (int v = 0; v < 8; ++v) {
        int q = qw + v + 8 * half;
        float inv = 1.0f / l_i[v];
        size_t row = ((size_t)b * SEQ + q) * (NH * HD) + h * HD;
#pragma unroll
        for (int nt = 0; nt < 4; ++nt)
            Out[row + nt * 16 + mlane] = (_Float16)(oacc[nt][v] * inv);
    }
}

// ---------------------------------------------------------------------------
extern "C" void kernel_launch(void* const* d_in, const int* in_sizes, int n_in,
                              void* d_out, int out_size, void* d_ws, size_t ws_size,
                              hipStream_t stream) {
    const float* x  = (const float*)d_in[0];
    const float* wq = (const float*)d_in[1];
    const float* wk = (const float*)d_in[2];
    const float* wv = (const float*)d_in[3];
    const float* wo = (const float*)d_in[4];

    const int M = 2 * SEQ;  // 4096
    char* ws = (char*)d_ws;
    const size_t MB = 1u << 20;
    _Float16* xh   = (_Float16*)(ws);             // [4096][2048]      16MB
    _Float16* wqT  = (_Float16*)(ws + 16 * MB);   // [2048][2048]       8MB
    _Float16* wkT  = (_Float16*)(ws + 24 * MB);   // [512][2048]        2MB
    _Float16* wvT  = (_Float16*)(ws + 26 * MB);   // [512][2048]        2MB
    _Float16* woT  = (_Float16*)(ws + 28 * MB);   // [2048][2048]       8MB
    _Float16* Qb   = (_Float16*)(ws + 36 * MB);   // [2][32][S][64]    16MB
    _Float16* Kbf  = (_Float16*)(ws + 52 * MB);   // [2][8][S][64]      4MB
    _Float16* VbfT = (_Float16*)(ws + 56 * MB);   // [2][8][64][S]      4MB
    _Float16* attn = xh;  // xh dead after the QKV GEMMs

    // precision conversion + weight transposes
    f32_to_f16<<<(M * DMODEL) / 256, 256, 0, stream>>>(x, xh);
    transpose_to_f16<<<dim3(DMODEL / 32, DMODEL / 32), dim3(32, 8), 0, stream>>>(wq, wqT, DMODEL, DMODEL);
    transpose_to_f16<<<dim3(DMODEL / 32, 512 / 32),    dim3(32, 8), 0, stream>>>(wk, wkT, DMODEL, 512);
    transpose_to_f16<<<dim3(DMODEL / 32, 512 / 32),    dim3(32, 8), 0, stream>>>(wv, wvT, DMODEL, 512);
    transpose_to_f16<<<dim3(DMODEL / 32, DMODEL / 32), dim3(32, 8), 0, stream>>>(wo, woT, DMODEL, DMODEL);

    // QKV projections with fused RoPE / relayout epilogues
    gemm_f16_async<EPI_ROPE><<<dim3(DMODEL / 128, M / 128), 256, 0, stream>>>(
        xh, wqT, Qb, M, DMODEL, DMODEL, NH);
    gemm_f16_async<EPI_ROPE><<<dim3(512 / 128, M / 128), 256, 0, stream>>>(
        xh, wkT, Kbf, M, 512, DMODEL, NKV);
    gemm_f16_async<EPI_COPY_T><<<dim3(512 / 128, M / 128), 256, 0, stream>>>(
        xh, wvT, VbfT, M, 512, DMODEL, NKV);

    // causal GQA flash attention
    flash_attn<<<2 * NH * (SEQ / 64), 128, 0, stream>>>(Qb, Kbf, VbfT, attn);

    // output projection -> f32
    gemm_f16_async<EPI_F32><<<dim3(DMODEL / 128, M / 128), 256, 0, stream>>>(
        attn, woT, d_out, M, DMODEL, DMODEL, 0);
}
